// EdgeFunc_15401752724192
// MI455X (gfx1250) — matrix operand order repaired and checked by reading, not compiled, not run on hardware
//
#include <hip/hip_runtime.h>

typedef _Float16 v16h __attribute__((ext_vector_type(16)));
typedef _Float16 v8h  __attribute__((ext_vector_type(8)));
typedef float    v8f  __attribute__((ext_vector_type(8)));
typedef float    f4   __attribute__((ext_vector_type(4)));

#define CH     128
#define EIN    384
#define W1KP   400            // padded K stride (halves) for W1t in ws/LDS: 800B rows
#define W2KP   144            // padded K stride (halves) for W2t: 288B rows
#define HPH    136            // padded channel stride (halves) for h1 f16 scratch
#define OUTP   132            // padded channel stride (floats) for out staging
#define MW     32             // edges per wave
#define EPS    1e-5f

#define W1T_HALFS (CH * W1KP)           // 51200
#define W2T_HALFS (CH * W2KP)           // 18432
#define SH_BYTES  (8 * MW * HPH * 2)    // 69632  (h1 f16 scratch, 8 waves)
#define WAVE_SLICE (MW * HPH * 2)       // 8704 bytes per wave
#define REGION_BYTES (SH_BYTES + W2T_HALFS * 2)   // 106496 > 102400 (W1 phase)

// raw v_rsq_f32: argument is >= EPS, never denormal, so skip libm's denorm guard
__device__ __forceinline__ float fast_rsqrt(float x) {
#if __has_builtin(__builtin_amdgcn_rsqf)
    return __builtin_amdgcn_rsqf(x);
#else
    return rsqrtf(x);
#endif
}

// ---------------- prep: W1[384,128], W2[128,128] f32 -> transposed f16 in ws ----------------
__global__ __launch_bounds__(256) void prep_weights(const float* __restrict__ W1,
                                                    const float* __restrict__ W2,
                                                    _Float16* __restrict__ wt) {
    int idx = blockIdx.x * 256 + threadIdx.x;
    if (idx < CH * EIN) {                       // w1t[n][k] = W1[k][n]
        int n = idx / EIN, k = idx - n * EIN;
        wt[n * W1KP + k] = (_Float16)W1[k * CH + n];
    }
    int i2 = idx - CH * EIN;
    if (i2 >= 0 && i2 < CH * CH) {              // w2t[n][k] = W2[k][n]
        int n = i2 / CH, k = i2 - n * CH;
        wt[W1T_HALFS + n * W2KP + k] = (_Float16)W2[k * CH + n];
    }
}

// GroupNorm(8) + affine + ReLU on 8 values held in two f4 regs -> one half of an A fragment
__device__ __forceinline__ void gn_pack(f4 x0, f4 x1,
                                        const float* __restrict__ gw,
                                        const float* __restrict__ gb,
                                        v16h& af, int half) {
    float s  = ((x0.x + x0.y) + (x0.z + x0.w)) + ((x1.x + x1.y) + (x1.z + x1.w));
    float ss = (x0.x*x0.x + x0.y*x0.y + x0.z*x0.z + x0.w*x0.w)
             + (x1.x*x1.x + x1.y*x1.y + x1.z*x1.z + x1.w*x1.w);
    float mu   = s * 0.125f;
    float var  = ss * 0.125f - mu * mu;
    float rstd = fast_rsqrt(var + EPS);
#pragma unroll
    for (int j = 0; j < 8; ++j) {
        float xv = (j < 4) ? x0[j] : x1[j - 4];
        float y  = (xv - mu) * rstd * gw[j] + gb[j];
        af[half * 8 + j] = (_Float16)fmaxf(y, 0.0f);
    }
}

// Same, but source is 8 contiguous f16 in LDS
__device__ __forceinline__ void gn_pack_h(const _Float16* __restrict__ p,
                                          const float* __restrict__ gw,
                                          const float* __restrict__ gb,
                                          v16h& af, int half) {
    v8h xh = *(const v8h*)p;
    float x[8];
#pragma unroll
    for (int j = 0; j < 8; ++j) x[j] = (float)xh[j];
    float s  = ((x[0] + x[1]) + (x[2] + x[3])) + ((x[4] + x[5]) + (x[6] + x[7]));
    float ss = (x[0]*x[0] + x[1]*x[1] + x[2]*x[2] + x[3]*x[3])
             + (x[4]*x[4] + x[5]*x[5] + x[6]*x[6] + x[7]*x[7]);
    float mu   = s * 0.125f;
    float var  = ss * 0.125f - mu * mu;
    float rstd = fast_rsqrt(var + EPS);
#pragma unroll
    for (int j = 0; j < 8; ++j) {
        float y = (x[j] - mu) * rstd * gw[j] + gb[j];
        af[half * 8 + j] = (_Float16)fmaxf(y, 0.0f);
    }
}

// ---------------- fused edge MLP ----------------
__global__ __launch_bounds__(256) void edge_mlp(
    const float* __restrict__ srcT, const float* __restrict__ tgtT,
    const float* __restrict__ eatT,
    const float* __restrict__ gn1w, const float* __restrict__ gn1b,
    const float* __restrict__ b1,
    const float* __restrict__ gn2w, const float* __restrict__ gn2b,
    const float* __restrict__ b2,
    const _Float16* __restrict__ wt,
    float* __restrict__ outp)
{
    // phase-overlapped LDS region: [phase1] W1t (102400B) | [phase2] h1-f16 (69632B) + W2t (36864B)
    __shared__ __align__(64) unsigned char smem[REGION_BYTES];
    __shared__ float sGn1w[EIN], sGn1b[EIN];
    __shared__ float sB1[CH], sGn2w[CH], sGn2b[CH], sB2[CH];

    _Float16* const sW1 = (_Float16*)smem;
    _Float16* const sHh = (_Float16*)smem;
    _Float16* const sW2 = (_Float16*)(smem + SH_BYTES);

    const int tid = threadIdx.x;

    // stage W1t + per-channel vectors
    {
        const uint32_t* w1s = (const uint32_t*)wt;
        uint32_t* d1 = (uint32_t*)sW1;
        for (int i = tid; i < W1T_HALFS / 2; i += 256) d1[i] = w1s[i];
        for (int i = tid; i < EIN; i += 256) { sGn1w[i] = gn1w[i]; sGn1b[i] = gn1b[i]; }
        if (tid < CH) {
            sB1[tid] = b1[tid]; sGn2w[tid] = gn2w[tid];
            sGn2b[tid] = gn2b[tid]; sB2[tid] = b2[tid];
        }
    }
    __syncthreads();

    const int wave = tid >> 5;
    const int lane = tid & 31;
    const int nl   = lane & 15;
    const int h    = lane >> 4;
    const int edge0 = blockIdx.x * 256 + wave * MW;

    v8f acc[2][8];
    const v8f vz = {};
#pragma unroll
    for (int t = 0; t < 2; ++t)
#pragma unroll
        for (int nt = 0; nt < 8; ++nt) acc[t][nt] = vz;

    // raw x loader for k-step kt (8 contiguous f32 per chunk, 2 chunks, 2 M-tiles)
    auto load_x = [&](int kt, f4 xr[2][4]) {
        const float* base; int cb;
        if (kt < 4)      { base = srcT; cb = kt * 32; }
        else if (kt < 8) { base = tgtT; cb = (kt - 4) * 32; }
        else             { base = eatT; cb = (kt - 8) * 32; }
#pragma unroll
        for (int t = 0; t < 2; ++t) {
            const float* p = base + (edge0 + t * 16 + nl) * CH + cb + 8 * h;
            xr[t][0] = *(const f4*)p;
            xr[t][1] = *(const f4*)(p + 4);
            xr[t][2] = *(const f4*)(p + 16);
            xr[t][3] = *(const f4*)(p + 20);
        }
    };

    // ---- phase 1: h1 = GN1(concat) @ W1, software-pipelined global loads ----
    f4 xc[2][4];
    load_x(0, xc);
#pragma unroll
    for (int kt = 0; kt < 12; ++kt) {
        f4 xn[2][4];
        if (kt < 11) load_x(kt + 1, xn);           // prefetch next k-step
        v16h af[2];
#pragma unroll
        for (int t = 0; t < 2; ++t)
#pragma unroll
            for (int ci = 0; ci < 2; ++ci) {
                int gc = kt * 32 + ci * 16 + 8 * h;
                gn_pack(xc[t][2 * ci], xc[t][2 * ci + 1], &sGn1w[gc], &sGn1b[gc], af[t], ci);
            }
        const int kb = kt * 32 + 8 * h;
#pragma unroll
        for (int nt = 0; nt < 8; ++nt) {
            v8h blo = *(const v8h*)&sW1[(nt * 16 + nl) * W1KP + kb]; v8h bhi = *(const v8h*)&sW1[(nt * 16 + nl) * W1KP + kb + 16]; v16h bf = __builtin_shufflevector(blo, bhi, 0, 1, 2, 3, 4, 5, 6, 7, 8, 9, 10, 11, 12, 13, 14, 15);
            acc[0][nt] = __builtin_amdgcn_wmma_f32_16x16x32_f16(false, af[0], false, bf,
                                                                (short)0, acc[0][nt], false, false);
            acc[1][nt] = __builtin_amdgcn_wmma_f32_16x16x32_f16(false, af[1], false, bf,
                                                                (short)0, acc[1][nt], false, false);
        }
#pragma unroll
        for (int t = 0; t < 2; ++t)
#pragma unroll
            for (int q = 0; q < 4; ++q) xc[t][q] = xn[t][q];
    }

    __syncthreads();   // everyone done reading sW1; region may be repurposed

    // ---- epilogue 1: +b1, spill h1 as f16; stage W2t cooperatively ----
    _Float16* hw = sHh + wave * (MW * HPH);
#pragma unroll
    for (int t = 0; t < 2; ++t)
#pragma unroll
        for (int nt = 0; nt < 8; ++nt) {
            int col = nt * 16 + nl;
            float bv = sB1[col];
#pragma unroll
            for (int j = 0; j < 8; ++j)
                hw[(t * 16 + 8 * h + j) * HPH + col] = (_Float16)(acc[t][nt][j] + bv);
        }
    {
        const uint32_t* w2s = (const uint32_t*)(wt + W1T_HALFS);
        uint32_t* d2 = (uint32_t*)sW2;
        for (int i = tid; i < W2T_HALFS / 2; i += 256) d2[i] = w2s[i];
    }
    __syncthreads();

    // ---- phase 2: out = GN2(h1).relu @ W2 ----
#pragma unroll
    for (int t = 0; t < 2; ++t)
#pragma unroll
        for (int nt = 0; nt < 8; ++nt) acc[t][nt] = vz;

#pragma unroll
    for (int kt = 0; kt < 4; ++kt) {
        v16h af[2];
#pragma unroll
        for (int t = 0; t < 2; ++t) {
            int rowm = t * 16 + nl;
#pragma unroll
            for (int ci = 0; ci < 2; ++ci) {
                int c = kt * 32 + ci * 16 + 8 * h;
                gn_pack_h(hw + rowm * HPH + c, &sGn2w[c], &sGn2b[c], af[t], ci);
            }
        }
        const int kb = kt * 32 + 8 * h;
#pragma unroll
        for (int nt = 0; nt < 8; ++nt) {
            v8h blo = *(const v8h*)&sW2[(nt * 16 + nl) * W2KP + kb]; v8h bhi = *(const v8h*)&sW2[(nt * 16 + nl) * W2KP + kb + 16]; v16h bf = __builtin_shufflevector(blo, bhi, 0, 1, 2, 3, 4, 5, 6, 7, 8, 9, 10, 11, 12, 13, 14, 15);
            acc[0][nt] = __builtin_amdgcn_wmma_f32_16x16x32_f16(false, af[0], false, bf,
                                                                (short)0, acc[0][nt], false, false);
            acc[1][nt] = __builtin_amdgcn_wmma_f32_16x16x32_f16(false, af[1], false, bf,
                                                                (short)0, acc[1][nt], false, false);
        }
    }
    asm volatile("" ::: "memory");

    // ---- epilogue 2: +b2 via LDS transpose, coalesced f4 store with residual ----
    // reuse this wave's (now dead) h1 slice as a 16x132 f32 staging tile
    float* ow = (float*)((unsigned char*)smem + wave * WAVE_SLICE);
#pragma unroll
    for (int t = 0; t < 2; ++t) {
#pragma unroll
        for (int nt = 0; nt < 8; ++nt) {
            int col = nt * 16 + nl;
            float bv = sB2[col];
#pragma unroll
            for (int j = 0; j < 8; ++j)
                ow[(8 * h + j) * OUTP + col] = acc[t][nt][j] + bv;
        }
        asm volatile("" ::: "memory");
        const int gbase = (edge0 + t * 16) * CH;   // 16 contiguous output rows
#pragma unroll
        for (int it = 0; it < 16; ++it) {          // one row (512B) per wave-iteration
            f4 v = *(const f4*)&ow[it * OUTP + lane * 4];
            f4 r = *(const f4*)&eatT[gbase + it * CH + lane * 4];
            *(f4*)&outp[gbase + it * CH + lane * 4] = v + r;
        }
        asm volatile("" ::: "memory");
    }
}

extern "C" void kernel_launch(void* const* d_in, const int* in_sizes, int n_in,
                              void* d_out, int out_size, void* d_ws, size_t ws_size,
                              hipStream_t stream) {
    const float* srcT  = (const float*)d_in[0];
    const float* tgtT  = (const float*)d_in[1];
    const float* eatT  = (const float*)d_in[2];
    const float* gn1w  = (const float*)d_in[3];
    const float* gn1b  = (const float*)d_in[4];
    const float* W1    = (const float*)d_in[5];
    const float* b1    = (const float*)d_in[6];
    const float* gn2w  = (const float*)d_in[7];
    const float* gn2b  = (const float*)d_in[8];
    const float* W2    = (const float*)d_in[9];
    const float* b2    = (const float*)d_in[10];
    float* out = (float*)d_out;
    _Float16* wt = (_Float16*)d_ws;   // needs (51200+18432)*2 = 139264 bytes

    prep_weights<<<(CH * EIN + CH * CH + 255) / 256, 256, 0, stream>>>(W1, W2, wt);

    const int E = in_sizes[0] / CH;   // 262144
    edge_mlp<<<E / 256, 256, 0, stream>>>(srcT, tgtT, eatT, gn1w, gn1b, b1,
                                          gn2w, gn2b, b2, wt, out);
}
